// CrossAttention_66082366816683
// MI455X (gfx1250) — matrix operand order repaired and checked by reading, not compiled, not run on hardware
//
#include <hip/hip_runtime.h>
#include <hip/hip_bf16.h>

#define DIMC 256
#define NPIX 3136
#define NB   8

typedef __bf16 bhalf;
typedef __attribute__((ext_vector_type(16))) __bf16 v16bf;
typedef __attribute__((ext_vector_type(8)))  __bf16 v8bf;
typedef __attribute__((ext_vector_type(8)))  float  v8f;
typedef __attribute__((ext_vector_type(4)))  float  v4f;

static __device__ __forceinline__ v8f wmma_bf16(v16bf a, v16bf b, v8f c) {
  return __builtin_amdgcn_wmma_f32_16x16x32_bf16(false, a, false, b, (short)0, c,
                                                 false, false);
}

// A tile (16 rows x K=32, bf16). Source is row-major with row stride `stride`
// elements; `base` already points at (row0, K0).
// Layout: lanes 0-15 -> row=l16, K in {0..7, 16..23}; lanes 16-31 -> row=l16,
// K in {8..15, 24..31}.
static __device__ __forceinline__ v16bf load_a(const bhalf* __restrict__ base,
                                               int stride, int l16, int hf) {
  const bhalf* p = base + (size_t)l16 * stride + hf * 8;
  union { v16bf v; v8bf h[2]; } u;
  u.h[0] = *(const v8bf*)(p);
  u.h[1] = *(const v8bf*)(p + 16);
  return u.v;
}

// B tile (K=32 x 16 cols, bf16). Source rows correspond to B columns; the
// contiguous dimension is K. Lanes 0-15: col=l16, K in {0..7, 16..23}; lanes
// 16-31: col=l16, K in {8..15, 24..31}.
static __device__ __forceinline__ v16bf load_b(const bhalf* __restrict__ base,
                                               size_t stride, int l16, int hf) {
  const bhalf* p = base + (size_t)l16 * stride + hf * 8;
  union { v16bf v; v8bf h[2]; } u;
  u.h[0] = *(const v8bf*)(p);
  u.h[1] = *(const v8bf*)(p + 16);
  return u.v;
}

__global__ void f32_to_bf16_kernel(const float* __restrict__ s,
                                   bhalf* __restrict__ d, int n) {
  int i = blockIdx.x * blockDim.x + threadIdx.x;
  if (i < n) d[i] = (bhalf)s[i];
}

__global__ void rden_kernel(const float* __restrict__ qn,
                            const float* __restrict__ kn,
                            float* __restrict__ rd, int n) {
  int i = blockIdx.x * blockDim.x + threadIdx.x;
  if (i < n) rd[i] = 1.0f / (qn[i] * kn[i] + 1e-6f);
}

// y = W @ x + b for one batch slice. x is (B, C, N) fp32; W is (256, C) bf16.
// MODE 0: write y row-major (B, N, 256) bf16 + per-position L2 norm.
// MODE 1: write y col-major (B, 256, N) bf16 (for V).
template <int MODE>
__global__ __launch_bounds__(128) void proj_x_kernel(
    const float* __restrict__ x, const bhalf* __restrict__ W,
    const float* __restrict__ bias, bhalf* __restrict__ yrow,
    bhalf* __restrict__ ycol, float* __restrict__ norm) {
  __shared__ bhalf sA[4][16][32];
  const int w = threadIdx.x >> 5, lane = threadIdx.x & 31;
  const int hf = lane >> 4, l16 = lane & 15;
  const int b = blockIdx.y;
  const int n0 = (blockIdx.x * 4 + w) * 16;

  // Build A panel: xT tile (16 n-rows x 256 c) in bf16 registers, transposing
  // 32c x 16n fp32 chunks through LDS.
  v16bf a[8];
#pragma unroll
  for (int kk = 0; kk < 8; ++kk) {
    const float* xp = x + ((size_t)b * DIMC + kk * 32 + lane) * NPIX + n0;
    v4f f0 = *(const v4f*)(xp);
    v4f f1 = *(const v4f*)(xp + 4);
    v4f f2 = *(const v4f*)(xp + 8);
    v4f f3 = *(const v4f*)(xp + 12);
    __syncthreads();  // WAR vs previous chunk's reads
    sA[w][0][lane] = (bhalf)f0.x;  sA[w][1][lane] = (bhalf)f0.y;
    sA[w][2][lane] = (bhalf)f0.z;  sA[w][3][lane] = (bhalf)f0.w;
    sA[w][4][lane] = (bhalf)f1.x;  sA[w][5][lane] = (bhalf)f1.y;
    sA[w][6][lane] = (bhalf)f1.z;  sA[w][7][lane] = (bhalf)f1.w;
    sA[w][8][lane] = (bhalf)f2.x;  sA[w][9][lane] = (bhalf)f2.y;
    sA[w][10][lane] = (bhalf)f2.z; sA[w][11][lane] = (bhalf)f2.w;
    sA[w][12][lane] = (bhalf)f3.x; sA[w][13][lane] = (bhalf)f3.y;
    sA[w][14][lane] = (bhalf)f3.z; sA[w][15][lane] = (bhalf)f3.w;
    __syncthreads();  // RAW
    a[kk] = load_a((const bhalf*)&sA[w][0][0], 32, l16, hf);
  }

  float sumsq[8];
#pragma unroll
  for (int r = 0; r < 8; ++r) sumsq[r] = 0.0f;

#pragma unroll 1
  for (int dt = 0; dt < 16; ++dt) {
    v8f acc = {0.f, 0.f, 0.f, 0.f, 0.f, 0.f, 0.f, 0.f};
#pragma unroll
    for (int kk = 0; kk < 8; ++kk)
      acc = wmma_bf16(a[kk], load_b(W + (size_t)(dt * 16) * DIMC + kk * 32,
                                    DIMC, l16, hf), acc);
    const float bv = bias[dt * 16 + l16];
    if (MODE == 0) {
#pragma unroll
      for (int r = 0; r < 8; ++r) {
        float q = acc[r] + bv;
        sumsq[r] += q * q;
        yrow[((size_t)b * NPIX + n0 + hf * 8 + r) * DIMC + dt * 16 + l16] =
            (bhalf)q;
      }
    } else {
      v8bf pk;
#pragma unroll
      for (int r = 0; r < 8; ++r) pk[r] = (bhalf)(acc[r] + bv);
      *(v8bf*)(ycol + ((size_t)b * DIMC + dt * 16 + l16) * NPIX + n0 + hf * 8) =
          pk;
    }
  }

  if (MODE == 0) {
#pragma unroll
    for (int r = 0; r < 8; ++r) {
      float s = sumsq[r];
      s += __shfl_xor(s, 1, 32);
      s += __shfl_xor(s, 2, 32);
      s += __shfl_xor(s, 4, 32);
      s += __shfl_xor(s, 8, 32);
      if (l16 == 0) norm[(size_t)b * NPIX + n0 + hf * 8 + r] = sqrtf(s);
    }
  }
}

// Flash attention: one wave owns 16 query rows; loops over keys in steps of 32.
__global__ __launch_bounds__(128) void flash_kernel(
    const bhalf* __restrict__ qT,   // (B, N, C)
    const bhalf* __restrict__ kT,   // (B, N, C)
    const bhalf* __restrict__ vC,   // (B, C, N)
    const float* __restrict__ rden, // (B, N)
    bhalf* __restrict__ oT) {       // (B, N, C)
  __shared__ bhalf sP[4][16][32];
  const int w = threadIdx.x >> 5, lane = threadIdx.x & 31;
  const int hf = lane >> 4, l16 = lane & 15;
  const int b = blockIdx.y;
  const int n0 = (blockIdx.x * 4 + w) * 16;

  v16bf aq[8];
#pragma unroll
  for (int kk = 0; kk < 8; ++kk)
    aq[kk] = load_a(qT + ((size_t)b * NPIX + n0) * DIMC + kk * 32, DIMC, l16, hf);

  v8f o[16];
#pragma unroll
  for (int ct = 0; ct < 16; ++ct)
#pragma unroll
    for (int r = 0; r < 8; ++r) o[ct][r] = 0.0f;
  float mr[8], lr[8];
#pragma unroll
  for (int r = 0; r < 8; ++r) { mr[r] = -3.0e38f; lr[r] = 0.0f; }

#pragma unroll 1
  for (int it = 0; it < NPIX / 32; ++it) {
    const int m0 = it * 32;
    const bhalf* kb0 = kT + ((size_t)b * NPIX + m0) * DIMC;
    const bhalf* kb1 = kb0 + 16 * DIMC;
    if (it + 1 < NPIX / 32)
      __builtin_prefetch(kb0 + 32 * DIMC + (size_t)lane * 64, 0, 1);

    v8f s0 = {0.f, 0.f, 0.f, 0.f, 0.f, 0.f, 0.f, 0.f};
    v8f s1 = {0.f, 0.f, 0.f, 0.f, 0.f, 0.f, 0.f, 0.f};
#pragma unroll
    for (int kk = 0; kk < 8; ++kk) {
      s0 = wmma_bf16(aq[kk], load_b(kb0 + kk * 32, DIMC, l16, hf), s0);
      s1 = wmma_bf16(aq[kk], load_b(kb1 + kk * 32, DIMC, l16, hf), s1);
    }

    // Reference: attn[n,m] scaled by 1/(qn[m]*kn[m]+eps)  (column-only scale)
    const float r0 = rden[(size_t)b * NPIX + m0 + l16];
    const float r1 = rden[(size_t)b * NPIX + m0 + 16 + l16];

    float p0[8], p1[8], corr[8];
#pragma unroll
    for (int r = 0; r < 8; ++r) {
      float t0 = s0[r] * r0, t1 = s1[r] * r1;
      float pm = fmaxf(t0, t1);
      pm = fmaxf(pm, __shfl_xor(pm, 1, 32));
      pm = fmaxf(pm, __shfl_xor(pm, 2, 32));
      pm = fmaxf(pm, __shfl_xor(pm, 4, 32));
      pm = fmaxf(pm, __shfl_xor(pm, 8, 32));
      float mn = fmaxf(mr[r], pm);
      float c = __expf(mr[r] - mn);
      float e0 = __expf(t0 - mn);
      float e1 = __expf(t1 - mn);
      float ps = e0 + e1;
      ps += __shfl_xor(ps, 1, 32);
      ps += __shfl_xor(ps, 2, 32);
      ps += __shfl_xor(ps, 4, 32);
      ps += __shfl_xor(ps, 8, 32);
      lr[r] = lr[r] * c + ps;
      mr[r] = mn;
      corr[r] = c; p0[r] = e0; p1[r] = e1;
    }
#pragma unroll
    for (int ct = 0; ct < 16; ++ct)
#pragma unroll
      for (int r = 0; r < 8; ++r) o[ct][r] *= corr[r];

    // Relayout P from C/D layout to A layout via per-wave LDS.
    __syncthreads();
#pragma unroll
    for (int r = 0; r < 8; ++r) {
      sP[w][hf * 8 + r][l16] = (bhalf)p0[r];
      sP[w][hf * 8 + r][16 + l16] = (bhalf)p1[r];
    }
    __syncthreads();
    v16bf ap = load_a((const bhalf*)&sP[w][0][0], 32, l16, hf);
#pragma unroll
    for (int ct = 0; ct < 16; ++ct)
      o[ct] = wmma_bf16(ap, load_b(vC + ((size_t)b * DIMC + ct * 16) * NPIX + m0,
                                   NPIX, l16, hf), o[ct]);
  }

  float inv[8];
#pragma unroll
  for (int r = 0; r < 8; ++r) inv[r] = 1.0f / lr[r];
#pragma unroll
  for (int ct = 0; ct < 16; ++ct)
#pragma unroll
    for (int r = 0; r < 8; ++r)
      oT[((size_t)b * NPIX + n0 + hf * 8 + r) * DIMC + ct * 16 + l16] =
          (bhalf)(o[ct][r] * inv[r]);
}

// out = Wproj @ o + bproj + x_q, written as (B, C, N) fp32.
__global__ __launch_bounds__(128) void proj_out_kernel(
    const bhalf* __restrict__ oT, const bhalf* __restrict__ W,
    const float* __restrict__ bias, const float* __restrict__ xres,
    float* __restrict__ out) {
  const int w = threadIdx.x >> 5, lane = threadIdx.x & 31;
  const int hf = lane >> 4, l16 = lane & 15;
  const int b = blockIdx.y;
  const int n0 = (blockIdx.x * 4 + w) * 16;

  v16bf a[8];
#pragma unroll
  for (int kk = 0; kk < 8; ++kk)
    a[kk] = load_a(oT + ((size_t)b * NPIX + n0) * DIMC + kk * 32, DIMC, l16, hf);

#pragma unroll 1
  for (int dt = 0; dt < 16; ++dt) {
    v8f acc = {0.f, 0.f, 0.f, 0.f, 0.f, 0.f, 0.f, 0.f};
#pragma unroll
    for (int kk = 0; kk < 8; ++kk)
      acc = wmma_bf16(a[kk], load_b(W + (size_t)(dt * 16) * DIMC + kk * 32,
                                    DIMC, l16, hf), acc);
    const int d = dt * 16 + l16;
    const float bv = bias[d];
    const size_t idx = ((size_t)b * DIMC + d) * NPIX + n0 + hf * 8;
    v8f xr = *(const v8f*)(xres + idx);
    v8f res;
#pragma unroll
    for (int r = 0; r < 8; ++r) res[r] = acc[r] + bv + xr[r];
    *(v8f*)(out + idx) = res;
  }
}

extern "C" void kernel_launch(void* const* d_in, const int* in_sizes, int n_in,
                              void* d_out, int out_size, void* d_ws,
                              size_t ws_size, hipStream_t stream) {
  const float* x_q  = (const float*)d_in[0];
  const float* x_kv = (const float*)d_in[1];
  const float* Wq   = (const float*)d_in[2];
  const float* bq   = (const float*)d_in[3];
  const float* Wkv  = (const float*)d_in[4];
  const float* bkv  = (const float*)d_in[5];
  const float* Wpr  = (const float*)d_in[6];
  const float* bpr  = (const float*)d_in[7];
  float* out = (float*)d_out;

  char* p = (char*)d_ws;
  auto take = [&](size_t bytes) {
    char* r = p;
    p += (bytes + 255) & ~(size_t)255;
    return r;
  };
  bhalf* wq  = (bhalf*)take((size_t)256 * 256 * 2);
  bhalf* wkv = (bhalf*)take((size_t)512 * 256 * 2);
  bhalf* wpr = (bhalf*)take((size_t)256 * 256 * 2);
  const size_t ebn = (size_t)NB * NPIX * DIMC;
  bhalf* qT = (bhalf*)take(ebn * 2);
  bhalf* kT = (bhalf*)take(ebn * 2);
  bhalf* vC = (bhalf*)take(ebn * 2);
  bhalf* oT = (bhalf*)take(ebn * 2);
  float* qn = (float*)take((size_t)NB * NPIX * 4);
  float* kn = (float*)take((size_t)NB * NPIX * 4);
  float* rd = (float*)take((size_t)NB * NPIX * 4);

  f32_to_bf16_kernel<<<(65536 + 255) / 256, 256, 0, stream>>>(Wq, wq, 65536);
  f32_to_bf16_kernel<<<(131072 + 255) / 256, 256, 0, stream>>>(Wkv, wkv, 131072);
  f32_to_bf16_kernel<<<(65536 + 255) / 256, 256, 0, stream>>>(Wpr, wpr, 65536);

  dim3 grid(49, NB), blk(128);
  proj_x_kernel<0><<<grid, blk, 0, stream>>>(x_q, wq, bq, qT, nullptr, qn);
  proj_x_kernel<0><<<grid, blk, 0, stream>>>(x_kv, wkv, bkv, kT, nullptr, kn);
  proj_x_kernel<1><<<grid, blk, 0, stream>>>(x_kv, wkv + (size_t)256 * 256,
                                             bkv + 256, nullptr, vC, nullptr);

  const int nt = NB * NPIX;
  rden_kernel<<<(nt + 255) / 256, 256, 0, stream>>>(qn, kn, rd, nt);
  flash_kernel<<<grid, blk, 0, stream>>>(qT, kT, vC, rd, oT);
  proj_out_kernel<<<grid, blk, 0, stream>>>(oT, wpr, bpr, x_q, out);
}
